// CausalWanSelfAttention_45140106281746
// MI455X (gfx1250) — matrix operand order repaired and checked by reading, not compiled, not run on hardware
//
#include <hip/hip_runtime.h>

// ---------------- problem constants ----------------
#define LSEQ   3120          // 2*30*52
#define LP     3136          // padded to 49*64
#define DIMD   1536
#define NHEAD  12
#define HDIM   128
#define FRAME  1560          // 30*52
#define SCALE_QK 0.08838834764831845f   // 1/sqrt(128)

typedef __attribute__((ext_vector_type(16))) __bf16 bf16x16;
typedef __attribute__((ext_vector_type(8)))  float  f32x8;

union BfOp { unsigned int u[8]; bf16x16 v; };

// -------- CDNA5 async global->LDS path (guarded; falls back to sync copy) ----
#define GLB_AS __attribute__((address_space(1)))
#define LDS_AS __attribute__((address_space(3)))

typedef int v4i_t __attribute__((vector_size(16)));   // matches builtin's V4i

#if defined(__has_builtin)
#  if __has_builtin(__builtin_amdgcn_global_load_async_to_lds_b128)
#    define HAVE_ASYNC_LDS 1
#  endif
#endif
#ifndef HAVE_ASYNC_LDS
#  define HAVE_ASYNC_LDS 0
#endif

__device__ __forceinline__ void async_copy16(const void* g, void* l) {
#if HAVE_ASYNC_LDS
  __builtin_amdgcn_global_load_async_to_lds_b128(
      (GLB_AS v4i_t*)(unsigned long long)g, (LDS_AS v4i_t*)l, 0, 0);
#else
  *(uint4*)l = *(const uint4*)g;
#endif
}

__device__ __forceinline__ void async_wait0() {
#if HAVE_ASYNC_LDS
#  if defined(__has_builtin) && __has_builtin(__builtin_amdgcn_s_wait_asynccnt)
  __builtin_amdgcn_s_wait_asynccnt(0);
#  else
  asm volatile("s_wait_asynccnt 0x0" ::: "memory");
#  endif
#endif
}

__device__ __forceinline__ unsigned short f2bf(float f) {
  unsigned int u = __float_as_uint(f);
  unsigned int r = u + 0x7FFFu + ((u >> 16) & 1u);
  return (unsigned short)(r >> 16);
}

__device__ __forceinline__ f32x8 zero8() {
  f32x8 z;
#pragma unroll
  for (int i = 0; i < 8; ++i) z[i] = 0.0f;
  return z;
}

// ---------------- prep kernels ----------------
__global__ __launch_bounds__(256) void prep_x_kernel(const float* __restrict__ x,
                                                     unsigned short* __restrict__ xb) {
  int i = blockIdx.x * 256 + threadIdx.x;
  if (i >= LP * DIMD) return;
  int row = i / DIMD;
  float v = (row < LSEQ) ? x[i] : 0.0f;
  xb[i] = f2bf(v);
}

// wt[w][n][k] = w_src[k][n]  (bf16, N-major so B-operand K-pairs are contiguous)
__global__ __launch_bounds__(256) void prep_w_kernel(const float* __restrict__ wq,
                                                     const float* __restrict__ wk,
                                                     const float* __restrict__ wv,
                                                     const float* __restrict__ wo,
                                                     unsigned short* __restrict__ wt) {
  int i = blockIdx.x * 256 + threadIdx.x;
  if (i >= 4 * DIMD * DIMD) return;
  int w   = i / (DIMD * DIMD);
  int rem = i % (DIMD * DIMD);
  int n = rem / DIMD, k = rem % DIMD;
  const float* src = (w == 0) ? wq : (w == 1) ? wk : (w == 2) ? wv : wo;
  wt[i] = f2bf(src[(size_t)k * DIMD + n]);
}

// vb[h][l][d] = bf16(vf[l][h*128+d])
__global__ __launch_bounds__(256) void conv_v_kernel(const float* __restrict__ vf,
                                                     unsigned short* __restrict__ vb) {
  int i = blockIdx.x * 256 + threadIdx.x;
  if (i >= LP * DIMD) return;
  int l = i / DIMD, c = i % DIMD;
  int h = c >> 7, d = c & 127;
  vb[((size_t)h * LP + l) * HDIM + d] = f2bf(vf[i]);
}

// ---------------- WMMA GEMM: out[M x 1536] = A(bf16) @ Bt^T + bias ----------------
// A: [LP][1536] bf16 row-major.  Bt: [1536 n][1536 k] bf16 (i.e. W transposed).
// Block: 256 thr (8 waves, 2x4), tile 64x128, K-step 32, double-buffered async LDS.
__global__ __launch_bounds__(256) void gemm_bf16_kernel(const unsigned short* __restrict__ A,
                                                        const unsigned short* __restrict__ Bt,
                                                        const float* __restrict__ bias,
                                                        float* __restrict__ out,
                                                        int Mstore) {
  __shared__ unsigned int As[2][64 * 16];    // 64 rows x 32 k (packed bf16 pairs)
  __shared__ unsigned int Bs[2][128 * 16];   // 128 n  x 32 k

  const int tid  = threadIdx.x;
  const int lane = tid & 31;
  const int wid  = tid >> 5;
  const int wm   = wid >> 2;        // 0..1
  const int wn   = wid & 3;         // 0..3
  const int half = lane >> 4;
  const int l16  = lane & 15;
  const int mBlk = blockIdx.x * 64;
  const int nBlk = blockIdx.y * 128;
  const int r    = tid >> 2;        // 0..63
  const int c4   = (tid & 3) * 4;   // uint offset within 16-uint row

  f32x8 acc[2][2];
#pragma unroll
  for (int a = 0; a < 2; ++a)
#pragma unroll
    for (int b = 0; b < 2; ++b) acc[a][b] = zero8();

  auto stage = [&](int buf, int kk) {
    async_copy16(A  + (size_t)(mBlk + r) * DIMD + kk + c4 * 2, &As[buf][r * 16 + c4]);
    async_copy16(Bt + (size_t)(nBlk + r) * DIMD + kk + c4 * 2, &Bs[buf][r * 16 + c4]);
    async_copy16(Bt + (size_t)(nBlk + 64 + r) * DIMD + kk + c4 * 2,
                 &Bs[buf][(64 + r) * 16 + c4]);
  };

  stage(0, 0);
  async_wait0();
  __syncthreads();

  for (int kk = 0; kk < DIMD; kk += 32) {
    const int buf = (kk >> 5) & 1;
    if (kk + 32 < DIMD) stage(buf ^ 1, kk + 32);   // fill other buffer asynchronously

    BfOp aop[2], bop[2];
#pragma unroll
    for (int mt = 0; mt < 2; ++mt) {
      const unsigned int* ap = &As[buf][(wm * 32 + mt * 16 + l16) * 16];
#pragma unroll
      for (int v = 0; v < 4; ++v) aop[mt].u[v]     = ap[half * 4 + v];       // K = half*8+2v
#pragma unroll
      for (int v = 0; v < 4; ++v) aop[mt].u[4 + v] = ap[8 + half * 4 + v];   // K = 16+half*8+2v
    }
#pragma unroll
    for (int nt = 0; nt < 2; ++nt) {
      const unsigned int* bp = &Bs[buf][(wn * 32 + nt * 16 + l16) * 16];
#pragma unroll
      for (int v = 0; v < 8; ++v) bop[nt].u[v] = bp[(v >> 2) * 8 + half * 4 + (v & 3)];   // K = (v>>2)*16+half*8+2(v&3)
    }
#pragma unroll
    for (int mt = 0; mt < 2; ++mt)
#pragma unroll
      for (int nt = 0; nt < 2; ++nt)
        acc[mt][nt] = __builtin_amdgcn_wmma_f32_16x16x32_bf16(
            false, aop[mt].v, false, bop[nt].v, (short)0, acc[mt][nt], false, false);

    async_wait0();     // next tile resident in LDS
    __syncthreads();   // all waves done reading current buffer
  }

  // epilogue: C/D layout -> row = r + 8*half, col = l16
#pragma unroll
  for (int mt = 0; mt < 2; ++mt)
#pragma unroll
    for (int nt = 0; nt < 2; ++nt) {
      int colg = nBlk + wn * 32 + nt * 16 + l16;
      float bv = bias[colg];
#pragma unroll
      for (int rr = 0; rr < 8; ++rr) {
        int rowg = mBlk + wm * 32 + mt * 16 + rr + half * 8;
        if (rowg < Mstore) out[(size_t)rowg * DIMD + colg] = acc[mt][nt][rr] + bv;
      }
    }
}

// ---------------- RMSNorm + RoPE (q / k), emits head-major bf16 ----------------
__global__ __launch_bounds__(256) void rmsrope_kernel(const float* __restrict__ qf,
                                                      const float* __restrict__ kf,
                                                      const float* __restrict__ gq,
                                                      const float* __restrict__ gk,
                                                      const float* __restrict__ fc,
                                                      const float* __restrict__ fs,
                                                      unsigned short* __restrict__ qb,
                                                      unsigned short* __restrict__ kb) {
  const int l = blockIdx.x;
  const int which = blockIdx.y;
  const float* in = ((which == 0) ? qf : kf) + (size_t)l * DIMD;
  const float* g  = (which == 0) ? gq : gk;
  unsigned short* ob = (which == 0) ? qb : kb;

  __shared__ float red[256];
  const int tid = threadIdx.x;
  float ss = 0.0f;
#pragma unroll
  for (int it = 0; it < 6; ++it) { float v = in[tid + it * 256]; ss += v * v; }
  red[tid] = ss;
  __syncthreads();
  for (int off = 128; off > 0; off >>= 1) {
    if (tid < off) red[tid] += red[tid + off];
    __syncthreads();
  }
  const float rms = rsqrtf(red[0] * (1.0f / (float)DIMD) + 1e-6f);

#pragma unroll
  for (int it = 0; it < 3; ++it) {
    int pp = tid + it * 256;         // pair index, 0..767
    int c0 = 2 * pp;
    int p  = pp & 63;                // pair within head
    float cv = 1.0f, sv = 0.0f;
    if (l < LSEQ) { cv = fc[l * 64 + p]; sv = fs[l * 64 + p]; }
    float y0 = in[c0] * rms * g[c0];
    float y1 = in[c0 + 1] * rms * g[c0 + 1];
    float o0 = y0 * cv - y1 * sv;
    float o1 = y0 * sv + y1 * cv;
    int hh = c0 >> 7, dl = c0 & 127;
    unsigned int pk = (unsigned int)f2bf(o0) | ((unsigned int)f2bf(o1) << 16);
    *((unsigned int*)&ob[((size_t)hh * LP + l) * HDIM + dl]) = pk;
  }
}

// ---------------- Flash attention (WMMA), per (64-query tile, head) ----------------
// Block: 128 thr = 4 waves; each wave owns 16 query rows. Key step = 32.
__global__ __launch_bounds__(128) void attn_kernel(const unsigned short* __restrict__ qb,
                                                   const unsigned short* __restrict__ kb,
                                                   const unsigned short* __restrict__ vb,
                                                   unsigned short* __restrict__ ab) {
  __shared__ unsigned int   Ks[32 * 64];      // 32 keys x 128 d (bf16 pairs)
  __shared__ unsigned int   Vt[128 * 16];     // 128 d x 32 j (transposed, bf16 pairs)
  __shared__ unsigned short Ps[4 * 16 * 32];  // per-wave P staging

  const int tid  = threadIdx.x;
  const int lane = tid & 31;
  const int wv   = tid >> 5;
  const int half = lane >> 4;
  const int l16  = lane & 15;
  const int h    = blockIdx.y;
  const int qBase = blockIdx.x * 64;

  const unsigned short* qh = qb + (size_t)h * LP * HDIM;
  const unsigned short* kh = kb + (size_t)h * LP * HDIM;
  const unsigned short* vh = vb + (size_t)h * LP * HDIM;

  // Q in registers, A-operand layout, 4 chunks of K=32 over HDIM
  BfOp qa[4];
  {
    const unsigned int* qrow =
        (const unsigned int*)(qh + (size_t)(qBase + wv * 16 + l16) * HDIM);
#pragma unroll
    for (int g = 0; g < 4; ++g) {
      int b = g * 16;
#pragma unroll
      for (int v = 0; v < 4; ++v) qa[g].u[v]     = qrow[b + half * 4 + v];
#pragma unroll
      for (int v = 0; v < 4; ++v) qa[g].u[4 + v] = qrow[b + 8 + half * 4 + v];
    }
  }

  int   fi[8];
  float mrow[8], lrow[8];
  f32x8 accv[8];
#pragma unroll
  for (int t = 0; t < 8; ++t) accv[t] = zero8();
#pragma unroll
  for (int r = 0; r < 8; ++r) {
    mrow[r] = -__builtin_inff();
    lrow[r] = 0.0f;
    fi[r] = (qBase + wv * 16 + r + half * 8) / FRAME;
  }

  const int kEnd = (((qBase + 63) / FRAME) == 0) ? 1568 : LP;  // 1568 = 49*32 >= 1560

  for (int j0 = 0; j0 < kEnd; j0 += 32) {
    __syncthreads();
    // stage K tile via async global->LDS: 32x128 bf16 = 512 uint4, 4 per thread
#pragma unroll
    for (int it = 0; it < 4; ++it) {
      int idx = tid + it * 128;
      int row = idx >> 4, c4 = idx & 15;
      async_copy16(kh + (size_t)(j0 + row) * HDIM + c4 * 8, &Ks[row * 64 + c4 * 4]);
    }
    // prefetch next V tile while async K loads are in flight
    if (j0 + 32 < kEnd)
      __builtin_prefetch(vh + (size_t)(j0 + 32) * HDIM + (size_t)tid * 32, 0, 0);
    // stage V transposed (manual: needs transpose): thread owns d = tid, packs j-pairs
    {
      int d = tid;
#pragma unroll
      for (int jp = 0; jp < 16; ++jp) {
        unsigned int lo = vh[(size_t)(j0 + 2 * jp) * HDIM + d];
        unsigned int hi = vh[(size_t)(j0 + 2 * jp + 1) * HDIM + d];
        Vt[d * 16 + jp] = lo | (hi << 16);
      }
    }
    async_wait0();
    __syncthreads();

    // S = Q @ K^T  (two 16-key column tiles)
    f32x8 s0 = zero8(), s1 = zero8();
#pragma unroll
    for (int g = 0; g < 4; ++g) {
      BfOp b0, b1;
      const unsigned int* k0 = &Ks[l16 * 64];
      const unsigned int* k1 = &Ks[(16 + l16) * 64];
#pragma unroll
      for (int v = 0; v < 8; ++v) {
        int ki = g * 16 + (v >> 2) * 8 + half * 4 + (v & 3);
        b0.u[v] = k0[ki];
        b1.u[v] = k1[ki];
      }
      s0 = __builtin_amdgcn_wmma_f32_16x16x32_bf16(false, qa[g].v, false, b0.v, (short)0, s0, false, false);
      s1 = __builtin_amdgcn_wmma_f32_16x16x32_bf16(false, qa[g].v, false, b1.v, (short)0, s1, false, false);
    }

    // scale + frame-causal mask + online softmax
    float alpha[8];
#pragma unroll
    for (int r = 0; r < 8; ++r) {
      float a0 = s0[r] * SCALE_QK, a1 = s1[r] * SCALE_QK;
      int jg0 = j0 + l16, jg1 = j0 + 16 + l16;
      if (jg0 >= LSEQ || (jg0 / FRAME) > fi[r]) a0 = -__builtin_inff();
      if (jg1 >= LSEQ || (jg1 / FRAME) > fi[r]) a1 = -__builtin_inff();
      float t = fmaxf(a0, a1);
      t = fmaxf(t, __shfl_xor(t, 1, 32));
      t = fmaxf(t, __shfl_xor(t, 2, 32));
      t = fmaxf(t, __shfl_xor(t, 4, 32));
      t = fmaxf(t, __shfl_xor(t, 8, 32));
      float mn = fmaxf(mrow[r], t);
      alpha[r] = __expf(mrow[r] - mn);
      mrow[r] = mn;
      float p0 = __expf(a0 - mn), p1 = __expf(a1 - mn);
      s0[r] = p0; s1[r] = p1;
      float rs = p0 + p1;
      rs += __shfl_xor(rs, 1, 32);
      rs += __shfl_xor(rs, 2, 32);
      rs += __shfl_xor(rs, 4, 32);
      rs += __shfl_xor(rs, 8, 32);
      lrow[r] = lrow[r] * alpha[r] + rs;
    }
#pragma unroll
    for (int t = 0; t < 8; ++t)
#pragma unroll
      for (int r = 0; r < 8; ++r) accv[t][r] *= alpha[r];

    // transpose P through LDS into A-operand layout (intra-wave; LDS is in-order)
#pragma unroll
    for (int r = 0; r < 8; ++r) {
      int prow = r + half * 8;
      Ps[(wv * 16 + prow) * 32 + l16]      = f2bf(s0[r]);
      Ps[(wv * 16 + prow) * 32 + 16 + l16] = f2bf(s1[r]);
    }
    BfOp pa;
    {
      const unsigned int* pr = (const unsigned int*)&Ps[(wv * 16 + l16) * 32];
#pragma unroll
      for (int v = 0; v < 4; ++v) pa.u[v]     = pr[half * 4 + v];
#pragma unroll
      for (int v = 0; v < 4; ++v) pa.u[4 + v] = pr[8 + half * 4 + v];
    }

    // O += P @ V  (8 output d-chunks of 16)
#pragma unroll
    for (int t = 0; t < 8; ++t) {
      BfOp bv_;
      const unsigned int* vp = &Vt[(t * 16 + l16) * 16];
#pragma unroll
      for (int v = 0; v < 8; ++v) bv_.u[v] = vp[(v >> 2) * 8 + half * 4 + (v & 3)];
      accv[t] = __builtin_amdgcn_wmma_f32_16x16x32_bf16(
          false, pa.v, false, bv_.v, (short)0, accv[t], false, false);
    }
  }

  // normalize + store bf16 to ab[l][h*128+d]
#pragma unroll
  for (int r = 0; r < 8; ++r) lrow[r] = 1.0f / lrow[r];
#pragma unroll
  for (int t = 0; t < 8; ++t)
#pragma unroll
    for (int r = 0; r < 8; ++r) {
      int rowg = qBase + wv * 16 + r + half * 8;
      int colg = h * HDIM + t * 16 + l16;
      ab[(size_t)rowg * DIMD + colg] = f2bf(accv[t][r] * lrow[r]);
    }
}

// ---------------- launch ----------------
extern "C" void kernel_launch(void* const* d_in, const int* in_sizes, int n_in,
                              void* d_out, int out_size, void* d_ws, size_t ws_size,
                              hipStream_t stream) {
  const float* x  = (const float*)d_in[0];
  const float* wq = (const float*)d_in[1];
  const float* wk = (const float*)d_in[2];
  const float* wv = (const float*)d_in[3];
  const float* wo = (const float*)d_in[4];
  const float* bq = (const float*)d_in[5];
  const float* bk = (const float*)d_in[6];
  const float* bv = (const float*)d_in[7];
  const float* bo = (const float*)d_in[8];
  const float* gq = (const float*)d_in[9];
  const float* gk = (const float*)d_in[10];
  const float* fc = (const float*)d_in[11];
  const float* fs = (const float*)d_in[12];
  float* outp = (float*)d_out;

  size_t off = 0;
  auto take = [&](size_t bytes) {
    void* p = (char*)d_ws + off;
    off += (bytes + 255) & ~(size_t)255;
    return p;
  };
  unsigned short* xb  = (unsigned short*)take((size_t)LP * DIMD * 2);
  unsigned short* wt  = (unsigned short*)take((size_t)4 * DIMD * DIMD * 2);
  float* qf = (float*)take((size_t)LP * DIMD * 4);
  float* kf = (float*)take((size_t)LP * DIMD * 4);
  float* vf = (float*)take((size_t)LP * DIMD * 4);
  unsigned short* qbh = (unsigned short*)take((size_t)LP * DIMD * 2);
  unsigned short* kbh = (unsigned short*)take((size_t)LP * DIMD * 2);
  unsigned short* vbh = (unsigned short*)take((size_t)LP * DIMD * 2);
  unsigned short* ab  = (unsigned short*)take((size_t)LP * DIMD * 2);
  (void)ws_size; (void)in_sizes; (void)n_in; (void)out_size;

  const int nElem = LP * DIMD;                 // 4,816,896 (divisible by 256)
  prep_x_kernel<<<nElem / 256, 256, 0, stream>>>(x, xb);
  prep_w_kernel<<<(4 * DIMD * DIMD) / 256, 256, 0, stream>>>(wq, wk, wv, wo, wt);

  dim3 gg(LP / 64, DIMD / 128);               // 49 x 12
  gemm_bf16_kernel<<<gg, 256, 0, stream>>>(xb, wt + (size_t)0 * DIMD * DIMD, bq, qf, LP);
  gemm_bf16_kernel<<<gg, 256, 0, stream>>>(xb, wt + (size_t)1 * DIMD * DIMD, bk, kf, LP);
  gemm_bf16_kernel<<<gg, 256, 0, stream>>>(xb, wt + (size_t)2 * DIMD * DIMD, bv, vf, LP);

  rmsrope_kernel<<<dim3(LP, 2), 256, 0, stream>>>(qf, kf, gq, gk, fc, fs, qbh, kbh);
  conv_v_kernel<<<nElem / 256, 256, 0, stream>>>(vf, vbh);

  attn_kernel<<<dim3(LP / 64, NHEAD), 128, 0, stream>>>(qbh, kbh, vbh, ab);

  gemm_bf16_kernel<<<gg, 256, 0, stream>>>(ab, wt + (size_t)3 * DIMD * DIMD, bo, outp, LSEQ);
}
